// CWVAE_61727269978386
// MI455X (gfx1250) — matrix operand order repaired and checked by reading, not compiled, not run on hardware
//
#include <hip/hip_runtime.h>
#include <cstdint>
#include <cstddef>

// ---------------------------------------------------------------------------
// CWVAE forward loss for MI455X (gfx1250, wave32, WMMA + TDM).
//   B=64 T=1024 D=E=H=256 Z=128, NL=2, TF={1,2}
// Strategy:
//   * all GEMMs via v_wmma_f32_16x16x32_f16 (f16 A/B from 0.02-scale f32, f32 acc)
//   * weights pre-transposed to [N x K] f16 once per call (B-fragment = 2x b128)
//   * recurrent scans: 1 persistent 1024-thread block, all state in LDS (~305KB)
//   * enc[t] tile (32KB contiguous) DMA'd by the Tensor Data Mover straight into
//     the strided `he` concat buffer using D# LDS padding (512B row + 512B pad),
//     overlapped with the prior-branch GEMMs; synced with s_wait_tensorcnt
//   * decoder: fused GEMM1->GEMM2->masked Gaussian log-prob per 16-row block
// ---------------------------------------------------------------------------

#define B_   64
#define T_   1024
#define D_   256
#define E_   256
#define H_   256
#define Z_   128
#define LOG2PI_F 1.8378770664093453f

typedef __attribute__((ext_vector_type(16))) _Float16 v16h;
typedef __attribute__((ext_vector_type(8)))  _Float16 v8h;
typedef __attribute__((ext_vector_type(8)))  float    v8f;

union AF16 { v16h v; v8h h[2]; };

enum { EP_RELU16 = 0, EP_RAW16 = 1, EP_RAW32 = 2, EP_SIG16 = 3 };

// ---- Tensor Data Mover availability (device pass only; guarded fallback) ----
#if defined(__HIP_DEVICE_COMPILE__) && defined(__gfx1250__) && \
    __has_builtin(__builtin_amdgcn_tensor_load_to_lds)
#define CWVAE_TDM 1
#else
#define CWVAE_TDM 0
#endif

#if CWVAE_TDM
typedef __attribute__((ext_vector_type(4))) unsigned int u32x4_t;
typedef __attribute__((ext_vector_type(8))) int          i32x8_t;
typedef __attribute__((ext_vector_type(4))) int          i32x4_t;

// DMA a [rows x width_halves] f16 tile (contiguous in global) into LDS at
// ldsByteOff, inserting `pad` DWORD gaps after every `interval` DWORDs
// (codes per CDNA5 D# group1: interval code p -> 2^(p+1) DWORDs,
//  amount code a -> a+1 DWORDs). padIntervalCode<0 disables padding.
__device__ __forceinline__ void tdm_load_tile_f16(
    const void* gsrc, unsigned ldsByteOff, int width, int rows,
    int padIntervalCode, int padAmountCode)
{
  unsigned long long ga = (unsigned long long)(uintptr_t)gsrc;
  unsigned w0g1 = (1u << 16);                       // data_size = 2 bytes
  if (padIntervalCode >= 0)
    w0g1 |= (1u << 20) | ((unsigned)padIntervalCode << 22)
                        | ((unsigned)padAmountCode << 25);
  const unsigned s1 = (unsigned)(width * rows);     // tensor_dim1_stride
  u32x4_t g0 = { 1u,                                // count=1, user mode
                 ldsByteOff,                        // lds_addr
                 (unsigned)ga,                      // global_addr[31:0]
                 ((unsigned)(ga >> 32) & 0x1FFFFFFu) | (2u << 30) }; // +type=2
  i32x8_t g1 = { (int)w0g1,
                 (int)(((unsigned)width & 0xFFFFu) << 16),           // dim0 lo16
                 (int)((((unsigned)width) >> 16) |
                       (((unsigned)rows & 0xFFFFu) << 16)),          // dim0 hi|dim1 lo
                 (int)((((unsigned)rows) >> 16) |
                       (((unsigned)width & 0xFFFFu) << 16)),         // dim1 hi|tile_dim0
                 (int)((unsigned)rows & 0xFFFFu),                    // tile_dim1
                 (int)(unsigned)width,                               // dim0_stride lo32
                 (int)((s1 & 0xFFFFu) << 16),                        // |dim1_stride lo16
                 (int)(s1 >> 16) };                                  // dim1_stride hi
  i32x4_t g2 = { 1, 1, 0, 0 };   // tensor_dim2=1, tensor_dim3=1
  i32x4_t g3 = { 0, 1 << 16, 0, 0 };  // tensor_dim4=1
#if __has_include(<hip/amd_detail/amd_gfx1250_TDM.h>)
  i32x8_t g4 = {};
  __builtin_amdgcn_tensor_load_to_lds(g0, g1, g2, g3, g4, 0);
#else
  __builtin_amdgcn_tensor_load_to_lds(g0, g1, g2, g3, 0);
#endif
}
#endif  // CWVAE_TDM

__device__ __forceinline__ void wait_tensorcnt0() {
#if CWVAE_TDM
#if __has_builtin(__builtin_amdgcn_s_wait_tensorcnt)
  __builtin_amdgcn_s_wait_tensorcnt(0);
#else
  asm volatile("s_wait_tensorcnt 0x0" ::: "memory");
#endif
#endif
}

__device__ __forceinline__ float softplus_f(float x) {
  return (x > 0.f) ? (x + log1pf(__expf(-x))) : log1pf(__expf(x));
}

// ---------------------------------------------------------------------------
// Wave-tiled WMMA GEMM:  OUT[M,N] = epilogue( A[M,K] @ Wt[N,K]^T (+A2@Wt2) + b )
// A, A2 : f16, row-major, lda = K (LDS or global)
// Wt    : f16, row-major N x K (pre-transposed weights, global)
// Fragment layouts per CDNA5 ISA 7.12.2 (16-bit A 16x32, 16-bit B 32x16).
// Must be called with uniform EXEC (all block threads).
// ---------------------------------------------------------------------------
__device__ __forceinline__ void gemm_f16(
    const _Float16* __restrict__ A,  const _Float16* __restrict__ Wt,  const float* __restrict__ bias,
    const _Float16* __restrict__ A2, const _Float16* __restrict__ Wt2, const float* __restrict__ bias2,
    int M, int K, int N, int ep,
    _Float16* __restrict__ out16, float* __restrict__ out32, int ldo)
{
  const int lane = (int)(threadIdx.x & 31u);
  const int wid  = (int)(threadIdx.x >> 5);
  const int nw   = (int)(blockDim.x >> 5);
  const int lm   = lane & 15;
  const int sel  = lane >> 4;           // 0: lanes 0-15, 1: lanes 16-31
  const int mtn  = M >> 4;
  const int ntn  = N >> 4;
  for (int tile = wid; tile < mtn * ntn; tile += nw) {
    const int mt = tile % mtn;
    const int nt = tile / mtn;
    const int am = mt * 16 + lm;        // A row held by this lane
    const int n  = nt * 16 + lm;        // B column / output column
    v8f acc = {};
    for (int k0 = 0; k0 < K; k0 += 32) {
      AF16 a, b;
      const _Float16* ap = A + (size_t)am * K + k0 + sel * 8;
      a.h[0] = *(const v8h*)(ap);       // K = k0+koff .. +7
      a.h[1] = *(const v8h*)(ap + 16);  // K = k0+16+koff .. +7
      const _Float16* bp = Wt + (size_t)n * K + k0 + sel * 8;
      b.h[0] = *(const v8h*)(bp);       // K = k0+sel*8 .. +7
      b.h[1] = *(const v8h*)(bp + 16);  // K = k0+16+sel*8 .. +7
      acc = __builtin_amdgcn_wmma_f32_16x16x32_f16(false, a.v, false, b.v,
                                                   (short)0, acc, false, false);
      if (A2) {
        AF16 a2, b2;
        const _Float16* ap2 = A2 + (size_t)am * K + k0 + sel * 8;
        a2.h[0] = *(const v8h*)(ap2);
        a2.h[1] = *(const v8h*)(ap2 + 16);
        const _Float16* bp2 = Wt2 + (size_t)n * K + k0 + sel * 8;
        b2.h[0] = *(const v8h*)(bp2);
        b2.h[1] = *(const v8h*)(bp2 + 16);
        acc = __builtin_amdgcn_wmma_f32_16x16x32_f16(false, a2.v, false, b2.v,
                                                     (short)0, acc, false, false);
      }
    }
    float bv = bias[n] + (bias2 ? bias2[n] : 0.f);
    const int mr0 = mt * 16 + sel * 8;  // C/D layout: VGPR v -> row v (+8 for hi lanes)
#pragma unroll
    for (int v = 0; v < 8; ++v) {
      float xv = acc[v] + bv;
      const int mr = mr0 + v;
      if (ep == EP_RELU16)      out16[(size_t)mr * ldo + n] = (_Float16)(xv > 0.f ? xv : 0.f);
      else if (ep == EP_RAW16)  out16[(size_t)mr * ldo + n] = (_Float16)xv;
      else if (ep == EP_RAW32)  out32[(size_t)mr * ldo + n] = xv;
      else { float s = 1.f / (1.f + __expf(-xv)); out16[(size_t)mr * ldo + n] = (_Float16)s; }
    }
  }
}

// GRU candidate-gate fused GEMM: two accumulators (gi@Wih[:,2H:], h@Whh[:,2H:]),
// epilogue applies n = tanh(in + r*hn); h' = (1-z)*n + z*h entirely in registers.
// rz: 64x512 f16 LDS (r cols 0..255, z cols 256..511). Writes hF (f32) only;
// the f16 copy of h is refreshed after a barrier (h16 is live GEMM input here).
__device__ __forceinline__ void gemm_gru(
    const _Float16* __restrict__ gi, const _Float16* __restrict__ h16,
    const _Float16* __restrict__ WihN, const float* __restrict__ bihN,
    const _Float16* __restrict__ WhhN, const float* __restrict__ bhhN,
    const _Float16* __restrict__ rz, float* __restrict__ hF)
{
  const int K = 256;
  const int lane = (int)(threadIdx.x & 31u);
  const int wid  = (int)(threadIdx.x >> 5);
  const int nw   = (int)(blockDim.x >> 5);
  const int lm = lane & 15, sel = lane >> 4;
  for (int tile = wid; tile < 4 * 16; tile += nw) {
    const int mt = tile & 3, nt = tile >> 2;
    const int am = mt * 16 + lm, n = nt * 16 + lm;
    v8f accI = {}, accH = {};
    for (int k0 = 0; k0 < K; k0 += 32) {
      AF16 a, b;
      const _Float16* ap = gi + (size_t)am * K + k0 + sel * 8;
      a.h[0] = *(const v8h*)ap; a.h[1] = *(const v8h*)(ap + 16);
      const _Float16* bp = WihN + (size_t)n * K + k0 + sel * 8;
      b.h[0] = *(const v8h*)bp; b.h[1] = *(const v8h*)(bp + 16);
      accI = __builtin_amdgcn_wmma_f32_16x16x32_f16(false, a.v, false, b.v,
                                                    (short)0, accI, false, false);
      const _Float16* ap2 = h16 + (size_t)am * K + k0 + sel * 8;
      a.h[0] = *(const v8h*)ap2; a.h[1] = *(const v8h*)(ap2 + 16);
      const _Float16* bp2 = WhhN + (size_t)n * K + k0 + sel * 8;
      b.h[0] = *(const v8h*)bp2; b.h[1] = *(const v8h*)(bp2 + 16);
      accH = __builtin_amdgcn_wmma_f32_16x16x32_f16(false, a.v, false, b.v,
                                                    (short)0, accH, false, false);
    }
    const float bI = bihN[n], bH = bhhN[n];
    const int mr0 = mt * 16 + sel * 8;
#pragma unroll
    for (int v = 0; v < 8; ++v) {
      const int mr = mr0 + v;
      float gin = accI[v] + bI;
      float ghn = accH[v] + bH;
      float r   = (float)rz[(size_t)mr * 512 + n];
      float zg  = (float)rz[(size_t)mr * 512 + 256 + n];
      float nn  = tanhf(gin + r * ghn);
      float hold = hF[(size_t)mr * 256 + n];
      hF[(size_t)mr * 256 + n] = (1.f - zg) * nn + zg * hold;
    }
  }
}

// ---------------------------------------------------------------------------
// Weight transpose+cast:  Wt[n*K + k] = (f16) W[k*N + n]
// ---------------------------------------------------------------------------
__global__ void k_castT(const float* __restrict__ W, _Float16* __restrict__ Wt, int K, int N) {
  int i = blockIdx.x * blockDim.x + threadIdx.x;
  if (i >= K * N) return;
  int n = i / K, k = i - n * K;
  Wt[i] = (_Float16)W[(size_t)k * N + n];
}

__global__ void k_zero(float* p, int n) {
  int i = blockIdx.x * blockDim.x + threadIdx.x;
  if (i < n) p[i] = 0.f;
}

// ---------------------------------------------------------------------------
// Encoder: pooled = mean_f(x); enc[t][b][:] = relu(pooled @ Wenc + b)  (f16 out)
// grid = (B/16, T/f), block = 256
// ---------------------------------------------------------------------------
__global__ __launch_bounds__(256) void k_encoder(
    const float* __restrict__ x, const _Float16* __restrict__ Wt,
    const float* __restrict__ bias, _Float16* __restrict__ enc, int f)
{
  __shared__ __align__(16) _Float16 A16[16 * 256];
  const int b0 = blockIdx.x * 16;
  const int tp = blockIdx.y;
  for (int i = threadIdx.x; i < 16 * 256; i += blockDim.x) {
    int row = i >> 8, d = i & 255;
    int b = b0 + row;
    float s = 0.f;
    for (int j = 0; j < f; ++j)
      s += x[((size_t)b * T_ + (size_t)tp * f + j) * D_ + d];
    A16[i] = (_Float16)(s / (float)f);
  }
  __syncthreads();
  _Float16* out = enc + ((size_t)tp * 64 + b0) * 256;   // [t][b][e] layout
  gemm_f16(A16, Wt, bias, nullptr, nullptr, nullptr,
           16, 256, 256, EP_RELU16, out, nullptr, 256);
}

// ---------------------------------------------------------------------------
// Persistent recurrent cell (one block, 32 waves). LDS carve (311,552 B):
//   hF(64x256 f32) qraw(64x256 f32) kacc(64) h16(64x256) z16(64x128)
//   S0(64x512) S1(64x256) praw16(64x256)
// ---------------------------------------------------------------------------
__global__ __launch_bounds__(1024) void k_cell(
    int Tsteps, int Kzc, int ctxShift, int tf,
    const _Float16* __restrict__ ctxIn,      // [Tin][64][384] or null
    const _Float16* __restrict__ enc,        // [Tsteps][64][256]
    const float* __restrict__ eps,           // [64][Tsteps][128]
    const int* __restrict__ x_sl,
    const _Float16* __restrict__ Wgt,  const float* __restrict__ bg,
    const _Float16* __restrict__ Wiht, const float* __restrict__ bih,
    const _Float16* __restrict__ Whht, const float* __restrict__ bhh,
    const _Float16* __restrict__ Wp1t, const float* __restrict__ bp1,
    const _Float16* __restrict__ Wp2t, const float* __restrict__ bp2,
    const _Float16* __restrict__ Wq1t, const float* __restrict__ bq1,
    const _Float16* __restrict__ Wq2t, const float* __restrict__ bq2,
    const _Float16* __restrict__ Wq3t, const float* __restrict__ bq3,
    _Float16* __restrict__ ctxOut,           // [Tsteps][64][384]
    float* __restrict__ kldAcc)              // [64]
{
  extern __shared__ __align__(16) char smem[];
  float*    hF     = (float*)smem;                 // 64*256 f32
  float*    qraw   = hF + 64 * 256;                // 64*256 f32
  float*    kacc   = qraw + 64 * 256;              // 64 f32
  _Float16* h16    = (_Float16*)(kacc + 64);       // 64*256
  _Float16* z16    = h16 + 64 * 256;               // 64*128
  _Float16* S0     = z16 + 64 * 128;               // 64*512
  _Float16* S1     = S0 + 64 * 512;                // 64*256
  _Float16* praw16 = S1 + 64 * 256;                // 64*256

  const int tid = (int)threadIdx.x;
  const int nth = (int)blockDim.x;
#if CWVAE_TDM
  // LDS byte offset of the enc half of `he` (S0 col 256) for the TDM D#.
  const unsigned ldsEncOff = (unsigned)__builtin_amdgcn_groupstaticsize() +
                             (unsigned)((const char*)S0 - (const char*)smem) + 512u;
#endif
  for (int i = tid; i < 64 * 256; i += nth) { hF[i] = 0.f; h16[i] = (_Float16)0.f; }
  for (int i = tid; i < 64 * 128; i += nth) z16[i] = (_Float16)0.f;
  if (tid < 64) kacc[tid] = 0.f;
  __syncthreads();

  for (int t = 0; t < Tsteps; ++t) {
    // hide L2 latency of next-step encoder rows (global_prefetch_b8)
    if (((tid & 127) == 0) && (t + 1 < Tsteps))
      __builtin_prefetch(enc + (size_t)(t + 1) * 64 * 256, 0, 2);

    // ---- zc = [z, ctx_t] ----
    const _Float16* Azc;
    if (Kzc > 128) {
      const _Float16* cin = ctxIn + (size_t)(t >> ctxShift) * 64 * 384;
      for (int i = tid; i < 64 * 512; i += nth) {
        int m = i >> 9, j = i & 511;
        S0[i] = (j < 128) ? z16[m * 128 + j] : cin[(size_t)m * 384 + (j - 128)];
      }
      Azc = S0;
      __syncthreads();
    } else {
      Azc = z16;   // level-1: ctx width 0
    }

    // gi = relu(zc @ Wg + bg) -> S1
    gemm_f16(Azc, Wgt, bg, nullptr, nullptr, nullptr, 64, Kzc, 256, EP_RELU16, S1, nullptr, 256);
    __syncthreads();

    // r,z gates = sigmoid(gi@Wih[:, :2H] + h@Whh[:, :2H] + bih + bhh) -> S0 (ld 512)
    gemm_f16(S1, Wiht, bih, h16, Whht, bhh, 64, 256, 512, EP_SIG16, S0, nullptr, 512);
    __syncthreads();

    // candidate + state update (writes hF)
    gemm_gru(S1, h16, Wiht + 512 * 256, bih + 512, Whht + 512 * 256, bhh + 512, S0, hF);
    __syncthreads();

#if CWVAE_TDM
    // rz (S0) fully consumed: DMA enc[t] (64 rows x 512B) into he's enc half,
    // padding 512B per row to skip the h half. Overlaps the p-branch GEMMs.
    if (tid < 32)
      tdm_load_tile_f16(enc + (size_t)t * 64 * 256, ldsEncOff,
                        256 /*halves/row*/, 64 /*rows*/,
                        6 /*pad every 128 DW*/, 127 /*pad 128 DW*/);
#endif

    for (int i = tid; i < 64 * 256; i += nth) h16[i] = (_Float16)hF[i];
    __syncthreads();

    // prior branch: p1 = relu(hn@Wp1+bp1); praw = p1@Wp2+bp2
    gemm_f16(h16, Wp1t, bp1, nullptr, nullptr, nullptr, 64, 256, 256, EP_RELU16, S1, nullptr, 256);
    __syncthreads();
    gemm_f16(S1, Wp2t, bp2, nullptr, nullptr, nullptr, 64, 256, 256, EP_RAW16, praw16, nullptr, 256);
    __syncthreads();

    // posterior branch: he = [hn, enc_t] -> S0
#if CWVAE_TDM
    for (int i = tid; i < 64 * 256; i += nth) {        // h half only; enc half via TDM
      int m = i >> 8, j = i & 255;
      S0[m * 512 + j] = h16[i];
    }
    if (tid < 32) wait_tensorcnt0();
#else
    {
      const _Float16* et = enc + (size_t)t * 64 * 256;
      for (int i = tid; i < 64 * 512; i += nth) {
        int m = i >> 9, j = i & 511;
        S0[i] = (j < 256) ? h16[m * 256 + j] : et[(size_t)m * 256 + (j - 256)];
      }
    }
#endif
    __syncthreads();
    gemm_f16(S0, Wq1t, bq1, nullptr, nullptr, nullptr, 64, 512, 256, EP_RELU16, S1, nullptr, 256);
    __syncthreads();
    gemm_f16(S1, Wq2t, bq2, nullptr, nullptr, nullptr, 64, 256, 256, EP_RELU16, S0, nullptr, 256);
    __syncthreads();
    gemm_f16(S0, Wq3t, bq3, nullptr, nullptr, nullptr, 64, 256, 256, EP_RAW32, nullptr, qraw, 256);
    __syncthreads();

    // z_n = qmu + qsd*eps ; masked KL accumulation
    for (int i = tid; i < 64 * 128; i += nth) {
      int m = i >> 7, j = i & 127;
      float qmu = qraw[m * 256 + j];
      float qs  = softplus_f(qraw[m * 256 + 128 + j]) + 1e-4f;
      float pmu = (float)praw16[m * 256 + j];
      float ps  = softplus_f((float)praw16[m * 256 + 128 + j]) + 1e-4f;
      float ev  = eps[((size_t)m * Tsteps + t) * 128 + j];
      z16[i] = (_Float16)(qmu + qs * ev);
      if (t * tf < x_sl[m]) {
        float dmu = qmu - pmu;
        float kl = logf(ps / qs) + (qs * qs + dmu * dmu) / (2.f * ps * ps) - 0.5f;
        atomicAdd(&kacc[m], kl);
      }
    }
    __syncthreads();

    // ctx_t = [zn, hn] -> global f16
    _Float16* co = ctxOut + (size_t)t * 64 * 384;
    for (int i = tid; i < 64 * 384; i += nth) {
      int m = i / 384, j = i - m * 384;
      co[i] = (j < 128) ? z16[m * 128 + j] : h16[m * 256 + (j - 128)];
    }
    __syncthreads();
  }
  if (tid < 64) kldAcc[tid] += kacc[tid];
}

// ---------------------------------------------------------------------------
// Decoder + masked Gaussian log-prob, fused per 16-row block (rows = t*64+b).
// grid = B*T/16 = 4096, block = 256.
// ---------------------------------------------------------------------------
__global__ __launch_bounds__(256) void k_decoder_loss(
    const _Float16* __restrict__ ctx0, const _Float16* __restrict__ W1t,
    const float* __restrict__ b1, const _Float16* __restrict__ W2t,
    const float* __restrict__ b2, const float* __restrict__ x,
    const int* __restrict__ x_sl, float* __restrict__ logprob)
{
  __shared__ __align__(16) _Float16 A16[16 * 384];
  __shared__ __align__(16) _Float16 d1[16 * 256];
  __shared__ __align__(16) float draw[16 * 512];
  __shared__ float rowsum[16];
  const int r0 = blockIdx.x * 16;
  const int t  = r0 >> 6;     // rows within a block share t (64 rows per t)
  const int b0 = r0 & 63;
  for (int i = threadIdx.x; i < 16 * 384; i += blockDim.x)
    A16[i] = ctx0[(size_t)r0 * 384 + i];
  if (threadIdx.x < 16) rowsum[threadIdx.x] = 0.f;
  __syncthreads();
  gemm_f16(A16, W1t, b1, nullptr, nullptr, nullptr, 16, 384, 256, EP_RELU16, d1, nullptr, 256);
  __syncthreads();
  gemm_f16(d1, W2t, b2, nullptr, nullptr, nullptr, 16, 256, 512, EP_RAW32, nullptr, draw, 512);
  __syncthreads();
  for (int i = threadIdx.x; i < 16 * 256; i += blockDim.x) {
    int row = i >> 8, d = i & 255;
    int b = b0 + row;
    if (t < x_sl[b]) {
      float mu = draw[row * 512 + d];
      float sd = softplus_f(draw[row * 512 + 256 + d]) + 1e-4f;
      float y  = x[((size_t)b * T_ + t) * D_ + d];
      float df = (y - mu) / sd;
      float lp = -0.5f * df * df - logf(sd) - 0.5f * LOG2PI_F;
      atomicAdd(&rowsum[row], lp);
    }
  }
  __syncthreads();
  if (threadIdx.x < 16)
    atomicAdd(&logprob[b0 + threadIdx.x], rowsum[threadIdx.x]);
}

// loss = -(sum_b(logprob - kld)) / sum_b(x_sl)
__global__ void k_final(const float* __restrict__ logprob, const float* __restrict__ kld,
                        const int* __restrict__ x_sl, float* __restrict__ out)
{
  __shared__ float s[64], ssl[64];
  int i = threadIdx.x;
  if (i < 64) { s[i] = logprob[i] - kld[i]; ssl[i] = (float)x_sl[i]; }
  __syncthreads();
  if (i == 0) {
    float a = 0.f, b = 0.f;
    for (int j = 0; j < 64; ++j) { a += s[j]; b += ssl[j]; }
    out[0] = -a / b;
  }
}

// ---------------------------------------------------------------------------
// Host side
// ---------------------------------------------------------------------------
// Input flattening assumption (setup_inputs dict order, params flattened in
// sorted-key DFS order as jax.tree_util does):
//   0:x 1:x_sl 2:eps0 3:eps1
//   4..19  cells[0]: Wg,Whh,Wih,Wp1,Wp2,Wq1,Wq2,Wq3,bg,bhh,bih,bp1,bp2,bq1,bq2,bq3
//   20..35 cells[1]: (same order)
//   36:dec_W1 37:dec_W2 38:dec_b1 39:dec_b2 40:enc_W0 41:enc_W1 42:enc_b0 43:enc_b1
enum { IW_Wg=0, IW_Whh, IW_Wih, IW_Wp1, IW_Wp2, IW_Wq1, IW_Wq2, IW_Wq3,
       IW_bg, IW_bhh, IW_bih, IW_bp1, IW_bp2, IW_bq1, IW_bq2, IW_bq3 };

extern "C" void kernel_launch(void* const* d_in, const int* in_sizes, int n_in,
                              void* d_out, int out_size, void* d_ws, size_t ws_size,
                              hipStream_t stream) {
  (void)in_sizes; (void)out_size; (void)ws_size;
  if (n_in < 44) return;

  const float* x    = (const float*)d_in[0];
  const int*   xsl  = (const int*)d_in[1];
  const float* eps0 = (const float*)d_in[2];
  const float* eps1 = (const float*)d_in[3];
  auto cellW = [&](int c, int which) -> const float* {
    return (const float*)d_in[4 + c * 16 + which];
  };
  const float* decW1 = (const float*)d_in[36];
  const float* decW2 = (const float*)d_in[37];
  const float* decb1 = (const float*)d_in[38];
  const float* decb2 = (const float*)d_in[39];
  const float* encW0 = (const float*)d_in[40];
  const float* encW1 = (const float*)d_in[41];
  const float* encb0 = (const float*)d_in[42];
  const float* encb1 = (const float*)d_in[43];

  // --- workspace carve (256B aligned) ---
  char* w = (char*)d_ws;
  auto alloc = [&](size_t bytes) -> char* {
    char* r = w; w += (bytes + 255) & ~(size_t)255; return r;
  };
  auto allocH = [&](size_t elems) -> _Float16* { return (_Float16*)alloc(elems * 2); };

  // transposed f16 weights (N x K)
  _Float16* WgT[2], *WihT[2], *WhhT[2], *Wp1T[2], *Wp2T[2], *Wq1T[2], *Wq2T[2], *Wq3T[2];
  const int KzcC[2] = { Z_ + (Z_ + H_), Z_ };   // cell0: 512, cell1: 128
  for (int c = 0; c < 2; ++c) {
    WgT[c]  = allocH((size_t)256 * KzcC[c]);
    WihT[c] = allocH((size_t)768 * 256);
    WhhT[c] = allocH((size_t)768 * 256);
    Wp1T[c] = allocH((size_t)256 * 256);
    Wp2T[c] = allocH((size_t)256 * 256);
    Wq1T[c] = allocH((size_t)256 * 512);
    Wq2T[c] = allocH((size_t)256 * 256);
    Wq3T[c] = allocH((size_t)256 * 256);
  }
  _Float16* dW1T = allocH((size_t)256 * 384);
  _Float16* dW2T = allocH((size_t)512 * 256);
  _Float16* eW0T = allocH((size_t)256 * 256);
  _Float16* eW1T = allocH((size_t)256 * 256);
  // activations
  _Float16* enc0 = allocH((size_t)T_ * 64 * 256);        // [t][b][e]
  _Float16* enc1 = allocH((size_t)(T_ / 2) * 64 * 256);
  _Float16* ctx1 = allocH((size_t)(T_ / 2) * 64 * 384);
  _Float16* ctx0 = allocH((size_t)T_ * 64 * 384);
  float* acc = (float*)alloc(128 * 4);                   // logprob[64], kld[64]
  float* logprob = acc;
  float* kldAcc  = acc + 64;

  auto castT = [&](const float* W, _Float16* Wt, int K, int N) {
    int n = K * N;
    k_castT<<<(n + 255) / 256, 256, 0, stream>>>(W, Wt, K, N);
  };
  for (int c = 0; c < 2; ++c) {
    castT(cellW(c, IW_Wg),  WgT[c],  KzcC[c], 256);
    castT(cellW(c, IW_Wih), WihT[c], 256, 768);
    castT(cellW(c, IW_Whh), WhhT[c], 256, 768);
    castT(cellW(c, IW_Wp1), Wp1T[c], 256, 256);
    castT(cellW(c, IW_Wp2), Wp2T[c], 256, 256);
    castT(cellW(c, IW_Wq1), Wq1T[c], 512, 256);
    castT(cellW(c, IW_Wq2), Wq2T[c], 256, 256);
    castT(cellW(c, IW_Wq3), Wq3T[c], 256, 256);
  }
  castT(decW1, dW1T, 384, 256);
  castT(decW2, dW2T, 256, 512);
  castT(encW0, eW0T, 256, 256);
  castT(encW1, eW1T, 256, 256);

  k_zero<<<1, 128, 0, stream>>>(acc, 128);

  // encoders
  {
    dim3 g0(4, T_), g1(4, T_ / 2);
    k_encoder<<<g0, 256, 0, stream>>>(x, eW0T, encb0, enc0, 1);
    k_encoder<<<g1, 256, 0, stream>>>(x, eW1T, encb1, enc1, 2);
  }

  const size_t CELL_SMEM =
      (size_t)(64 * 256 + 64 * 256 + 64) * 4 +
      (size_t)(64 * 256 + 64 * 128 + 64 * 512 + 64 * 256 + 64 * 256) * 2;  // 311,552 B

  // level 1 (top): 512 steps, no ctx input
  k_cell<<<1, 1024, CELL_SMEM, stream>>>(
      T_ / 2, Z_, 0, 2, nullptr, enc1, eps1, xsl,
      WgT[1], cellW(1, IW_bg), WihT[1], cellW(1, IW_bih), WhhT[1], cellW(1, IW_bhh),
      Wp1T[1], cellW(1, IW_bp1), Wp2T[1], cellW(1, IW_bp2),
      Wq1T[1], cellW(1, IW_bq1), Wq2T[1], cellW(1, IW_bq2), Wq3T[1], cellW(1, IW_bq3),
      ctx1, kldAcc);

  // level 0: 1024 steps, ctx from level 1 repeated x2 (t>>1)
  k_cell<<<1, 1024, CELL_SMEM, stream>>>(
      T_, KzcC[0], 1, 1, ctx1, enc0, eps0, xsl,
      WgT[0], cellW(0, IW_bg), WihT[0], cellW(0, IW_bih), WhhT[0], cellW(0, IW_bhh),
      Wp1T[0], cellW(0, IW_bp1), Wp2T[0], cellW(0, IW_bp2),
      Wq1T[0], cellW(0, IW_bq1), Wq2T[0], cellW(0, IW_bq2), Wq3T[0], cellW(0, IW_bq3),
      ctx0, kldAcc);

  // decoder + log-prob
  k_decoder_loss<<<(B_ * T_) / 16, 256, 0, stream>>>(
      ctx0, dW1T, decb1, dW2T, decb2, x, xsl, logprob);

  k_final<<<1, 64, 0, stream>>>(logprob, kldAcc, xsl, (float*)d_out);
}
